// RNN_21938692948140
// MI455X (gfx1250) — matrix operand order repaired and checked by reading, not compiled, not run on hardware
//
#include <hip/hip_runtime.h>
#include <hip/hip_bf16.h>
#include <math.h>

// Problem dims (match the reference)
constexpr int Bsz = 4096;
constexpr int Tst = 50;
constexpr int Idim = 64;
constexpr int Hdim = 512;
constexpr int G4H  = 4 * Hdim;  // 2048
constexpr int Odim = 8;

typedef __attribute__((ext_vector_type(8)))  _Float16 v8h;
typedef __attribute__((ext_vector_type(16))) _Float16 v16h;
typedef __attribute__((ext_vector_type(8)))  float    v8f;

// ---------------------------------------------------------------------------
// WMMA helpers (CDNA5 wave32, v_wmma_f32_16x16x32_f16)
// ---------------------------------------------------------------------------
__device__ __forceinline__ v16h pack16(v8h lo, v8h hi) {
  v16h r;
#pragma unroll
  for (int i = 0; i < 8; ++i) { r[i] = lo[i]; r[i + 8] = hi[i]; }
  return r;
}

// A-matrix 16x32 (MxK) f16 tile from row-major [*, ld] source.
// ISA layout: lane<16 -> M=lane, K chunks {0..7, 16..23}; lane>=16 ->
// M=lane-16, K chunks {8..15, 24..31}.  Two 16B loads per lane.
__device__ __forceinline__ v16h load_A(const _Float16* __restrict__ A, int ld,
                                       int m0, int k0, int lane) {
  const int m  = m0 + (lane & 15);
  const int kb = k0 + ((lane >> 4) << 3);
  const _Float16* p = A + (size_t)m * (size_t)ld + kb;
  v8h lo = *(const v8h*)(p);        // K = kb .. kb+7
  v8h hi = *(const v8h*)(p + 16);   // K = kb+16 .. kb+23
  return pack16(lo, hi);
}

// B-matrix 32x16 (KxN) f16 tile representing W^T, where W is [N x K]
// row-major.  ISA layout: lane<16 -> col N=lane, K chunks {0..7, 16..23};
// lane>=16 -> col N=lane-16, K chunks {8..15, 24..31}.  Two 16B loads.
__device__ __forceinline__ v16h load_Bt(const _Float16* __restrict__ W, int ld,
                                        int nrow, int k0, int lane) {
  const int n  = nrow + (lane & 15);
  const int kb = k0 + ((lane >> 4) << 3);
  const _Float16* p = W + (size_t)n * (size_t)ld + kb;
  v8h lo = *(const v8h*)(p);
  v8h hi = *(const v8h*)(p + 16);
  return pack16(lo, hi);
}

__device__ __forceinline__ v8f wmma_f16(v16h a, v16h b, v8f c) {
  return __builtin_amdgcn_wmma_f32_16x16x32_f16(
      /*neg_a=*/false, a, /*neg_b=*/false, b,
      /*c_mod=*/(short)0, c, /*reuse_a=*/false, /*reuse_b=*/false);
}

__device__ __forceinline__ float sigmoidf_(float x) {
  return 1.0f / (1.0f + __expf(-x));
}

// ---------------------------------------------------------------------------
// Fused LSTM step: gates = x_t @ Wih^T + h_prev @ Whh^T + bias ; cell update.
// Block = 256 thr (8 waves).  Each wave owns 32 rows x 16 h-cols and computes
// all 4 gate tiles (8 WMMA accumulators) so the B tiles are amortized over
// two M-subtiles: 12 b128 loads per 8 WMMAs.
// Block tile: 256 rows x 16 h-cols.  grid = (Hdim/16, Bsz/256).
// ---------------------------------------------------------------------------
__global__ __launch_bounds__(256) void lstm_step_kernel(
    const _Float16* __restrict__ xin,    // [Bsz, K_in] (row stride x_ld)
    int x_ld, int K_in,
    const _Float16* __restrict__ Wih,    // [4H, K_in] row-major f16
    const _Float16* __restrict__ Whh,    // [4H, H]    row-major f16
    const float*    __restrict__ bias,   // [4H]  (b_ih + b_hh)
    const _Float16* __restrict__ h_prev, // [Bsz, H] f16
    _Float16*       __restrict__ h_next, // [Bsz, H] f16
    float*          __restrict__ c)      // [Bsz, H] f32, updated in place
{
  const int lane = threadIdx.x & 31;
  const int wave = threadIdx.x >> 5;
  const int n0   = blockIdx.x << 4;                  // h-column base, 0..511
  const int m0   = (blockIdx.y << 8) + (wave << 5);  // batch-row base (32/wave)

  v8f a00 = {}, a01 = {}, a02 = {}, a03 = {};  // rows m0..m0+15, gates i,f,g,o
  v8f a10 = {}, a11 = {}, a12 = {}, a13 = {};  // rows m0+16..m0+31

  // Recurrent part: h_prev [Bsz x H] @ Whh^T
#pragma unroll 2
  for (int k0 = 0; k0 < Hdim; k0 += 32) {
    v16h a0 = load_A(h_prev, Hdim, m0,      k0, lane);
    v16h a1 = load_A(h_prev, Hdim, m0 + 16, k0, lane);
    v16h b0 = load_Bt(Whh, Hdim, 0 * Hdim + n0, k0, lane);
    v16h b1 = load_Bt(Whh, Hdim, 1 * Hdim + n0, k0, lane);
    v16h b2 = load_Bt(Whh, Hdim, 2 * Hdim + n0, k0, lane);
    v16h b3 = load_Bt(Whh, Hdim, 3 * Hdim + n0, k0, lane);
    a00 = wmma_f16(a0, b0, a00);  a10 = wmma_f16(a1, b0, a10);
    a01 = wmma_f16(a0, b1, a01);  a11 = wmma_f16(a1, b1, a11);
    a02 = wmma_f16(a0, b2, a02);  a12 = wmma_f16(a1, b2, a12);
    a03 = wmma_f16(a0, b3, a03);  a13 = wmma_f16(a1, b3, a13);
  }

  // Input part: x_t [Bsz x K_in] @ Wih^T
  for (int k0 = 0; k0 < K_in; k0 += 32) {
    v16h a0 = load_A(xin, x_ld, m0,      k0, lane);
    v16h a1 = load_A(xin, x_ld, m0 + 16, k0, lane);
    v16h b0 = load_Bt(Wih, K_in, 0 * Hdim + n0, k0, lane);
    v16h b1 = load_Bt(Wih, K_in, 1 * Hdim + n0, k0, lane);
    v16h b2 = load_Bt(Wih, K_in, 2 * Hdim + n0, k0, lane);
    v16h b3 = load_Bt(Wih, K_in, 3 * Hdim + n0, k0, lane);
    a00 = wmma_f16(a0, b0, a00);  a10 = wmma_f16(a1, b0, a10);
    a01 = wmma_f16(a0, b1, a01);  a11 = wmma_f16(a1, b1, a11);
    a02 = wmma_f16(a0, b2, a02);  a12 = wmma_f16(a1, b2, a12);
    a03 = wmma_f16(a0, b3, a03);  a13 = wmma_f16(a1, b3, a13);
  }

  // Epilogue: biases + LSTM cell update.
  // C/D layout: VGPR r -> row base + r + (lane<16 ? 0 : 8), col n0+(lane&15).
  const int col = n0 + (lane & 15);
  const float bi = bias[0 * Hdim + col];
  const float bf = bias[1 * Hdim + col];
  const float bg = bias[2 * Hdim + col];
  const float bo = bias[3 * Hdim + col];

#pragma unroll
  for (int s = 0; s < 2; ++s) {
    const v8f gi8 = s ? a10 : a00;
    const v8f gf8 = s ? a11 : a01;
    const v8f gg8 = s ? a12 : a02;
    const v8f go8 = s ? a13 : a03;
    const int rbase = m0 + 16 * s + ((lane >> 4) << 3);
#pragma unroll
    for (int r = 0; r < 8; ++r) {
      const size_t idx = (size_t)(rbase + r) * Hdim + col;
      const float gi = gi8[r] + bi;
      const float gf = gf8[r] + bf;
      const float gg = gg8[r] + bg;
      const float go = go8[r] + bo;
      const float cv = c[idx];
      const float cn = sigmoidf_(gf) * cv + sigmoidf_(gi) * tanhf(gg);
      const float hn = sigmoidf_(go) * tanhf(cn);
      c[idx] = cn;
      h_next[idx] = (_Float16)hn;
    }
  }
}

// ---------------------------------------------------------------------------
// Utility kernels
// ---------------------------------------------------------------------------
__global__ void f32_to_f16_kernel(const float* __restrict__ in,
                                  _Float16* __restrict__ out, int n) {
  int i = blockIdx.x * 256 + threadIdx.x;
  if (i < n) out[i] = (_Float16)in[i];
}

__global__ void bias_combine_kernel(const float* __restrict__ a,
                                    const float* __restrict__ b,
                                    float* __restrict__ out, int n) {
  int i = blockIdx.x * 256 + threadIdx.x;
  if (i < n) out[i] = a[i] + b[i];
}

// out[B, O] = h_last(f16) @ W_out^T(f32) + b_out ; tiny (O=8), VALU is fine.
__global__ void out_proj_kernel(const _Float16* __restrict__ h,
                                const float* __restrict__ Wout,
                                const float* __restrict__ bout,
                                float* __restrict__ out) {
  int idx = blockIdx.x * 256 + threadIdx.x;
  if (idx >= Bsz * Odim) return;
  int b = idx >> 3;
  int o = idx & 7;
  const _Float16* hp = h + (size_t)b * Hdim;
  const float* wp = Wout + (size_t)o * Hdim;
  float s = bout[o];
#pragma unroll 8
  for (int k = 0; k < Hdim; ++k) s += (float)hp[k] * wp[k];
  out[idx] = s;
}

// ---------------------------------------------------------------------------
// Host side
// ---------------------------------------------------------------------------
extern "C" void kernel_launch(void* const* d_in, const int* in_sizes, int n_in,
                              void* d_out, int out_size, void* d_ws, size_t ws_size,
                              hipStream_t stream) {
  (void)in_sizes; (void)n_in; (void)out_size; (void)ws_size;

  const float* x     = (const float*)d_in[0];   // [B,T,I]
  const float* Wih0  = (const float*)d_in[1];   // [4H, I]
  const float* Whh0  = (const float*)d_in[2];   // [4H, H]
  const float* bih0  = (const float*)d_in[3];   // [4H]
  const float* bhh0  = (const float*)d_in[4];   // [4H]
  const float* Wih1  = (const float*)d_in[5];   // [4H, H]
  const float* Whh1  = (const float*)d_in[6];   // [4H, H]
  const float* bih1  = (const float*)d_in[7];   // [4H]
  const float* bhh1  = (const float*)d_in[8];   // [4H]
  const float* Wout  = (const float*)d_in[9];   // [O, H]
  const float* bout  = (const float*)d_in[10];  // [O]
  float* out = (float*)d_out;

  // Workspace carve-out (256B aligned)
  char* ws = (char*)d_ws;
  size_t off = 0;
  auto carve = [&](size_t bytes) -> void* {
    void* p = ws + off;
    off = (off + bytes + 255) & ~(size_t)255;
    return p;
  };
  _Float16* xh    = (_Float16*)carve((size_t)Bsz * Tst * Idim * 2);
  _Float16* Wih0h = (_Float16*)carve((size_t)G4H * Idim * 2);
  _Float16* Whh0h = (_Float16*)carve((size_t)G4H * Hdim * 2);
  _Float16* Wih1h = (_Float16*)carve((size_t)G4H * Hdim * 2);
  _Float16* Whh1h = (_Float16*)carve((size_t)G4H * Hdim * 2);
  float*    bias0 = (float*)   carve((size_t)G4H * 4);
  float*    bias1 = (float*)   carve((size_t)G4H * 4);
  _Float16* h0a   = (_Float16*)carve((size_t)Bsz * Hdim * 2);
  _Float16* h0b   = (_Float16*)carve((size_t)Bsz * Hdim * 2);
  _Float16* h1a   = (_Float16*)carve((size_t)Bsz * Hdim * 2);
  _Float16* h1b   = (_Float16*)carve((size_t)Bsz * Hdim * 2);
  float*    c0    = (float*)   carve((size_t)Bsz * Hdim * 4);
  float*    c1    = (float*)   carve((size_t)Bsz * Hdim * 4);

  auto cvt = [&](const float* src, _Float16* dst, int n) {
    f32_to_f16_kernel<<<(n + 255) / 256, 256, 0, stream>>>(src, dst, n);
  };
  cvt(x,    xh,    Bsz * Tst * Idim);
  cvt(Wih0, Wih0h, G4H * Idim);
  cvt(Whh0, Whh0h, G4H * Hdim);
  cvt(Wih1, Wih1h, G4H * Hdim);
  cvt(Whh1, Whh1h, G4H * Hdim);
  bias_combine_kernel<<<(G4H + 255) / 256, 256, 0, stream>>>(bih0, bhh0, bias0, G4H);
  bias_combine_kernel<<<(G4H + 255) / 256, 256, 0, stream>>>(bih1, bhh1, bias1, G4H);

  // Zero initial states (graph-capturable async memsets)
  hipMemsetAsync(h0a, 0, (size_t)Bsz * Hdim * 2, stream);
  hipMemsetAsync(h1a, 0, (size_t)Bsz * Hdim * 2, stream);
  hipMemsetAsync(c0,  0, (size_t)Bsz * Hdim * 4, stream);
  hipMemsetAsync(c1,  0, (size_t)Bsz * Hdim * 4, stream);

  const dim3 grid(Hdim / 16, Bsz / 256);   // 32 x 16 blocks
  const dim3 block(256);

  _Float16* h1_last = h1a;
  for (int t = 0; t < Tst; ++t) {
    const _Float16* h0p = (t & 1) ? h0b : h0a;
    _Float16*       h0n = (t & 1) ? h0a : h0b;
    const _Float16* h1p = (t & 1) ? h1b : h1a;
    _Float16*       h1n = (t & 1) ? h1a : h1b;

    // Layer 0: input = x[:, t, :]  (row stride T*I)
    lstm_step_kernel<<<grid, block, 0, stream>>>(
        xh + (size_t)t * Idim, Tst * Idim, Idim,
        Wih0h, Whh0h, bias0, h0p, h0n, c0);

    // Layer 1: input = h0 at this timestep
    lstm_step_kernel<<<grid, block, 0, stream>>>(
        h0n, Hdim, Hdim,
        Wih1h, Whh1h, bias1, h1p, h1n, c1);

    h1_last = h1n;
  }

  out_proj_kernel<<<(Bsz * Odim + 255) / 256, 256, 0, stream>>>(
      h1_last, Wout, bout, out);
}
